// cosFormerAttention_75728863363115
// MI455X (gfx1250) — matrix operand order repaired and checked by reading, not compiled, not run on hardware
//
#include <hip/hip_runtime.h>

typedef __attribute__((ext_vector_type(16))) _Float16 v16h;
typedef __attribute__((ext_vector_type(8)))  _Float16 v8h;
typedef __attribute__((ext_vector_type(8)))  float    v8f;
typedef __attribute__((ext_vector_type(4)))  float    v4f;

#define SLEN   4096
#define DDIM   64
#define HNUM   8
#define BHNUM  32
#define NINTER 16
#define KCH    32     // s-rows per KV chunk
#define PT     40     // transposed f16 row stride (32 + pad) = 80 B, 16B-aligned
#define PQ     72     // f16 row stride (64 + pad) = 144 B, 16B-aligned
#define HALF_PI_F 1.57079632679489662f

__device__ __forceinline__ v16h hcat(v8h lo, v8h hi) {
    return __builtin_shufflevector(lo, hi, 0, 1, 2, 3, 4, 5, 6, 7,
                                           8, 9, 10, 11, 12, 13, 14, 15);
}

// ---------------------------------------------------------------------------
// Kernel 1: kv_sin/kv_cos = k_feat^T @ V, ksum = sum_s k_feat.
// Grid: (8 chunks of 512 rows, 32 bh). Block: 256 threads = 8 waves.
// Waves 0-3: sin matrix, d-tiles 0-3.  Waves 4-7: cos matrix, d-tiles 0-3.
// Feature tiles staged TRANSPOSED (s-minor): every WMMA fragment is two
// aligned ds_load_b128.
// ---------------------------------------------------------------------------
__global__ __launch_bounds__(256) void cosformer_kv_kernel(
    const float* __restrict__ Kin, const float* __restrict__ Vin,
    const unsigned char* __restrict__ maskp,
    const float* __restrict__ W1, const float* __restrict__ b1,
    const float* __restrict__ W2, const float* __restrict__ b2,
    float* __restrict__ kv, float* __restrict__ ksum)
{
    __shared__ __align__(16) float    sW1T[NINTER][DDIM];  // transposed weights
    __shared__ __align__(16) float    sW2[NINTER];
    __shared__ float    sb1[NINTER];
    __shared__ __align__(16) float    sKf[KCH][DDIM];      // relu+masked K (f32)
    __shared__ __align__(16) _Float16 sKsT[DDIM][PT];      // [d][s] sin features
    __shared__ __align__(16) _Float16 sKcT[DDIM][PT];      // [d][s] cos features
    __shared__ __align__(16) _Float16 sVT [DDIM][PT];      // [e][s] V
    __shared__ __align__(16) float    sH[KCH][NINTER];
    __shared__ float    sSin[KCH];
    __shared__ float    sCos[KCH];

    const int tid  = threadIdx.x;
    const int bh   = blockIdx.y;
    const int srowBase = blockIdx.x * 512;
    const float b2s = b2[0];
    const size_t rowBaseIdx = (size_t)bh * SLEN;

    for (int j = tid; j < DDIM * NINTER; j += 256)
        sW1T[j & 15][j >> 4] = W1[j];                      // W1[d][i] -> sW1T[i][d]
    if (tid < NINTER) { sW2[tid] = W2[tid]; sb1[tid] = b1[tid]; }

    const int lane = tid & 31;
    const int wave = tid >> 5;
    const int mat  = wave >> 2;   // 0 = sin, 1 = cos
    const int mt   = wave & 3;    // d-tile
    const int m    = lane & 15;
    const int hh   = lane >> 4;

    const int ccol = tid & 63;         // column for transpose / ksum tasks
    const int r0   = (tid >> 6) * 8;   // row group base (0,8,16,24)

    // hoisted per-thread base pointers (advance by constant per chunk)
    const v4f* K4              = (const v4f*)(Kin + (rowBaseIdx + srowBase) * DDIM) + tid;
    const float* vptr          = Vin + (rowBaseIdx + srowBase + r0) * DDIM + ccol;
    const unsigned char* mrow4 = maskp + (size_t)(bh / HNUM) * SLEN + srowBase + (tid >> 4);

    v8f acc[4] = {};
    float regSum = 0.0f;

    for (int t = 0; t < 16; ++t) {
        // ---- stage K rows (float4 coalesced, relu + mask) ------------------
        {
            #pragma unroll
            for (int i = 0; i < 2; ++i) {
                int idx = tid + i * 256;          // 512 float4 = 32x64
                int r = idx >> 4, c4 = idx & 15;
                v4f kx = K4[i * 256];
                float msk = mrow4[i * 16] ? 1.0f : 0.0f;
                v4f o;
                #pragma unroll
                for (int e = 0; e < 4; ++e) o[e] = fmaxf(kx[e], 0.0f) * msk;
                *(v4f*)&sKf[r][c4 * 4] = o;
            }
        }
        // ---- stage V transposed: one address, 8 offset-immediate loads -----
        {
            v8h hv;
            #pragma unroll
            for (int i = 0; i < 8; ++i)
                hv[i] = (_Float16)vptr[i * DDIM];
            *(v8h*)&sVT[ccol][r0] = hv;
        }
        if (t < 15) {  // prefetch next chunk (global_prefetch_b8)
            __builtin_prefetch(&K4[512], 0, 0);
            __builtin_prefetch(vptr + KCH * DDIM, 0, 0);
        }
        __syncthreads();
        // ---- MLP hidden: row = tid>>3, 2 of 16 outputs each ----------------
        {
            int r = tid >> 3, i0 = (tid & 7) * 2;
            float a0 = sb1[i0], a1 = sb1[i0 + 1];
            const v4f* xp = (const v4f*)sKf[r];
            const v4f* w0 = (const v4f*)sW1T[i0];
            const v4f* w1 = (const v4f*)sW1T[i0 + 1];
            #pragma unroll
            for (int d4 = 0; d4 < 16; ++d4) {
                v4f x = xp[d4], wa = w0[d4], wb = w1[d4];
                #pragma unroll
                for (int e = 0; e < 4; ++e) { a0 += x[e] * wa[e]; a1 += x[e] * wb[e]; }
            }
            sH[r][i0]     = fmaxf(a0, 0.0f);
            sH[r][i0 + 1] = fmaxf(a1, 0.0f);
        }
        __syncthreads();
        // ---- sigmoid head + sin/cos ----------------------------------------
        if (tid < KCH) {
            float z = b2s;
            const v4f* hp = (const v4f*)sH[tid];
            const v4f* wp = (const v4f*)sW2;
            #pragma unroll
            for (int i4 = 0; i4 < 4; ++i4) {
                v4f hx = hp[i4], wx = wp[i4];
                #pragma unroll
                for (int e = 0; e < 4; ++e) z += hx[e] * wx[e];
            }
            float p   = 1.0f / (1.0f + __expf(-z));
            float ang = HALF_PI_F * p;
            sSin[tid] = __sinf(ang);
            sCos[tid] = __cosf(ang);
        }
        __syncthreads();
        // ---- build transposed sin/cos feature tiles (one b128 store each) --
        {
            v8h hs, hc;
            #pragma unroll
            for (int i = 0; i < 8; ++i) {
                float kvl = sKf[r0 + i][ccol];
                hs[i] = (_Float16)(sSin[r0 + i] * kvl);
                hc[i] = (_Float16)(sCos[r0 + i] * kvl);
            }
            *(v8h*)&sKsT[ccol][r0] = hs;
            *(v8h*)&sKcT[ccol][r0] = hc;
        }
        __syncthreads();
        // ---- ksum column partials (vector LDS loads) -----------------------
        if (tid < 128) {
            const v8h* cp = (const v8h*)((tid < 64) ? sKsT[ccol] : sKcT[ccol]);
            #pragma unroll
            for (int g = 0; g < 4; ++g) {
                v8h x = cp[g];
                #pragma unroll
                for (int e = 0; e < 8; ++e) regSum += (float)x[e];
            }
        }
        // ---- WMMA: fragments = two aligned b128 LDS loads each -------------
        {
            _Float16 (*sA)[PT] = mat ? sKcT : sKsT;
            v16h a = hcat(*(const v8h*)&sA[mt * 16 + m][hh * 8],
                          *(const v8h*)&sA[mt * 16 + m][16 + hh * 8]);
            #pragma unroll
            for (int nt = 0; nt < 4; ++nt) {
                v16h bfr = hcat(*(const v8h*)&sVT[nt * 16 + m][hh * 8],
                                *(const v8h*)&sVT[nt * 16 + m][16 + hh * 8]);
                acc[nt] = __builtin_amdgcn_wmma_f32_16x16x32_f16(
                    false, a, false, bfr, (short)0, acc[nt], false, false);
            }
        }
        __syncthreads();
        K4    += 512;          // 32 rows * 16 float4
        vptr  += KCH * DDIM;
        mrow4 += KCH;
    }

    // ---- flush partials with f32 global atomics (one base + imm offsets) ---
    {
        float* kvb = kv + (((size_t)bh * 2 + mat) * DDIM + mt * 16 + hh * 8) * DDIM + m;
        #pragma unroll
        for (int nt = 0; nt < 4; ++nt) {
            #pragma unroll
            for (int r = 0; r < 8; ++r)
                atomicAdd(kvb + r * DDIM + nt * 16, acc[nt][r]);
        }
    }
    if (tid < 128)
        atomicAdd(&ksum[((size_t)bh * 2 + (tid >> 6)) * DDIM + (tid & 63)], regSum);
}

// ---------------------------------------------------------------------------
// Kernel 2: out = (q_sin@kv_sin + q_cos@kv_cos)/norm per 64-row block.
// Grid: (64 s-blocks, 32 bh). Block: 256 threads = 8 waves.
// Wave w: row-tile mt = w&3, n-tiles {2*(w>>2), 2*(w>>2)+1}.
// kv staged transposed [e][d] so B fragments are contiguous b128 loads.
// ---------------------------------------------------------------------------
__global__ __launch_bounds__(256) void cosformer_apply_kernel(
    const float* __restrict__ Qin,
    const float* __restrict__ W1, const float* __restrict__ b1,
    const float* __restrict__ W2, const float* __restrict__ b2,
    const float* __restrict__ kv, const float* __restrict__ ksum,
    float* __restrict__ out)
{
    __shared__ __align__(16) float    sW1T[NINTER][DDIM];
    __shared__ __align__(16) float    sW2[NINTER];
    __shared__ float    sb1[NINTER];
    __shared__ __align__(16) _Float16 sKVT[2][DDIM][PQ];   // [mat][e][d]
    __shared__ __align__(16) float    sKsum[2][DDIM];
    __shared__ __align__(16) _Float16 sQraw[64][PQ];       // relu(Q) f16
    __shared__ __align__(16) _Float16 sQs[64][PQ];
    __shared__ __align__(16) _Float16 sQc[64][PQ];
    __shared__ __align__(16) float    sH[64][NINTER];
    __shared__ float    sSin[64], sCos[64];
    __shared__ __align__(32) float    sNorm[64];

    const int tid = threadIdx.x;
    const int bh  = blockIdx.y;
    const int s0  = blockIdx.x * 64;
    const float b2s = b2[0];
    const size_t rowBaseIdx = (size_t)bh * SLEN;

    for (int j = tid; j < DDIM * NINTER; j += 256)
        sW1T[j & 15][j >> 4] = W1[j];
    if (tid < NINTER) { sW2[tid] = W2[tid]; sb1[tid] = b1[tid]; }
    // ---- stage kv transposed (f32 -> f16), coalesced on e ------------------
    {
        #pragma unroll
        for (int k = 0; k < 4; ++k) {
            int task = tid + k * 256;         // 1024 tasks = (mat, dg, e)
            int matk = task >> 9;
            int rem  = task & 511;
            int e    = rem & 63;
            int dg   = rem >> 6;
            const float* kvp = kv + (size_t)bh * 2 * DDIM * DDIM
                             + (size_t)matk * DDIM * DDIM + dg * 8 * DDIM + e;
            v8h hv;
            #pragma unroll
            for (int i = 0; i < 8; ++i)
                hv[i] = (_Float16)kvp[i * DDIM];
            *(v8h*)&sKVT[matk][e][dg * 8] = hv;
        }
    }
    if (tid < 128) sKsum[tid >> 6][tid & 63] = ksum[(size_t)bh * 2 * DDIM + tid];
    // ---- stage Q rows (float4 loads, relu, f16, b128 store) ----------------
    {
        const v4f* Q4 = (const v4f*)(Qin + (rowBaseIdx + s0) * DDIM) + tid * 2;
        #pragma unroll
        for (int k = 0; k < 2; ++k) {
            int task = tid + k * 256;         // 512 tasks = (row, col-group-of-8)
            int r = task >> 3, cg = task & 7;
            v4f x0 = Q4[k * 512];
            v4f x1 = Q4[k * 512 + 1];
            v8h h;
            #pragma unroll
            for (int e = 0; e < 4; ++e) {
                h[e]     = (_Float16)fmaxf(x0[e], 0.0f);
                h[4 + e] = (_Float16)fmaxf(x1[e], 0.0f);
            }
            *(v8h*)&sQraw[r][cg * 8] = h;
        }
    }
    __syncthreads();
    // ---- MLP hidden: 512 tasks, 2 outputs each -----------------------------
    #pragma unroll
    for (int k = 0; k < 2; ++k) {
        int task = tid + k * 256;
        int r = task >> 3, i0 = (task & 7) * 2;
        float a0 = sb1[i0], a1 = sb1[i0 + 1];
        const v8h* xp = (const v8h*)sQraw[r];
        const v4f* w0 = (const v4f*)sW1T[i0];
        const v4f* w1 = (const v4f*)sW1T[i0 + 1];
        #pragma unroll
        for (int d8 = 0; d8 < 8; ++d8) {
            v8h x8 = xp[d8];
            v4f wa0 = w0[d8 * 2], wa1 = w0[d8 * 2 + 1];
            v4f wb0 = w1[d8 * 2], wb1 = w1[d8 * 2 + 1];
            #pragma unroll
            for (int e = 0; e < 4; ++e) {
                a0 += (float)x8[e] * wa0[e];     a1 += (float)x8[e] * wb0[e];
                a0 += (float)x8[4 + e] * wa1[e]; a1 += (float)x8[4 + e] * wb1[e];
            }
        }
        sH[r][i0]     = fmaxf(a0, 0.0f);
        sH[r][i0 + 1] = fmaxf(a1, 0.0f);
    }
    __syncthreads();
    if (tid < 64) {
        float z = b2s;
        const v4f* hp = (const v4f*)sH[tid];
        const v4f* wp = (const v4f*)sW2;
        #pragma unroll
        for (int i4 = 0; i4 < 4; ++i4) {
            v4f hx = hp[i4], wx = wp[i4];
            #pragma unroll
            for (int e = 0; e < 4; ++e) z += hx[e] * wx[e];
        }
        float p   = 1.0f / (1.0f + __expf(-z));
        float ang = HALF_PI_F * p;
        sSin[tid] = __sinf(ang);
        sCos[tid] = __cosf(ang);
    }
    __syncthreads();
    // ---- build q_sin / q_cos tiles (vector LDS ops) ------------------------
    #pragma unroll
    for (int k = 0; k < 2; ++k) {
        int task = tid + k * 256;
        int r = task >> 3, cg = task & 7;
        v8h x = *(const v8h*)&sQraw[r][cg * 8];
        float sn = sSin[r], cs = sCos[r];
        v8h hs, hc;
        #pragma unroll
        for (int e = 0; e < 8; ++e) {
            float q = (float)x[e];
            hs[e] = (_Float16)(sn * q);
            hc[e] = (_Float16)(cs * q);
        }
        *(v8h*)&sQs[r][cg * 8] = hs;
        *(v8h*)&sQc[r][cg * 8] = hc;
    }
    __syncthreads();
    // ---- normalizer --------------------------------------------------------
    if (tid < 64) {
        float nacc = 0.0f;
        const v8h* qs = (const v8h*)sQs[tid];
        const v8h* qc = (const v8h*)sQc[tid];
        #pragma unroll
        for (int d8 = 0; d8 < 8; ++d8) {
            v8h a = qs[d8], b = qc[d8];
            #pragma unroll
            for (int e = 0; e < 8; ++e)
                nacc += (float)a[e] * sKsum[0][d8 * 8 + e]
                      + (float)b[e] * sKsum[1][d8 * 8 + e];
        }
        sNorm[tid] = nacc;
    }
    __syncthreads();

    const int lane = tid & 31;
    const int wave = tid >> 5;
    const int mt   = wave & 3;
    const int nh   = wave >> 2;
    const int m    = lane & 15;
    const int hh   = lane >> 4;
    const int rowA = mt * 16 + m;

    v8f acc[2] = {};
    #pragma unroll
    for (int k0 = 0; k0 < DDIM; k0 += 32) {
        v16h aS = hcat(*(const v8h*)&sQs[rowA][k0 + hh * 8],
                       *(const v8h*)&sQs[rowA][k0 + 16 + hh * 8]);
        v16h aC = hcat(*(const v8h*)&sQc[rowA][k0 + hh * 8],
                       *(const v8h*)&sQc[rowA][k0 + 16 + hh * 8]);
        #pragma unroll
        for (int q = 0; q < 2; ++q) {
            int nt = nh * 2 + q;
            v16h bS = hcat(*(const v8h*)&sKVT[0][nt * 16 + m][k0 + hh * 8],
                           *(const v8h*)&sKVT[0][nt * 16 + m][k0 + 16 + hh * 8]);
            v16h bC = hcat(*(const v8h*)&sKVT[1][nt * 16 + m][k0 + hh * 8],
                           *(const v8h*)&sKVT[1][nt * 16 + m][k0 + 16 + hh * 8]);
            acc[q] = __builtin_amdgcn_wmma_f32_16x16x32_f16(
                false, aS, false, bS, (short)0, acc[q], false, false);
            acc[q] = __builtin_amdgcn_wmma_f32_16x16x32_f16(
                false, aC, false, bC, (short)0, acc[q], false, false);
        }
    }

    // ---- reciprocal of norm once per row group, then write out -------------
    const int rowBase = mt * 16 + hh * 8;
    v8f nrm = *(const v8f*)&sNorm[rowBase];
    v8f inv;
    #pragma unroll
    for (int r = 0; r < 8; ++r) inv[r] = 1.0f / nrm[r];

    float* ob = out + (rowBaseIdx + s0 + rowBase) * DDIM + nh * 32 + m;
    #pragma unroll
    for (int q = 0; q < 2; ++q) {
        #pragma unroll
        for (int r = 0; r < 8; ++r)
            ob[r * DDIM + q * 16] = acc[q][r] * inv[r];
    }
}

// ---------------------------------------------------------------------------
extern "C" void kernel_launch(void* const* d_in, const int* in_sizes, int n_in,
                              void* d_out, int out_size, void* d_ws, size_t ws_size,
                              hipStream_t stream) {
    (void)in_sizes; (void)n_in; (void)out_size; (void)ws_size;
    const float* Q = (const float*)d_in[0];
    const float* K = (const float*)d_in[1];
    const float* V = (const float*)d_in[2];
    const unsigned char* mask = (const unsigned char*)d_in[3];
    const float* qW1 = (const float*)d_in[4];
    const float* qb1 = (const float*)d_in[5];
    const float* qW2 = (const float*)d_in[6];
    const float* qb2 = (const float*)d_in[7];
    const float* kW1 = (const float*)d_in[8];
    const float* kb1 = (const float*)d_in[9];
    const float* kW2 = (const float*)d_in[10];
    const float* kb2 = (const float*)d_in[11];

    float* kv = (float*)d_ws;                                   // [32][2][64][64]
    float* ks = kv + (size_t)BHNUM * 2 * DDIM * DDIM;           // [32][2][64]
    size_t zeroBytes = ((size_t)BHNUM * 2 * DDIM * DDIM +
                        (size_t)BHNUM * 2 * DDIM) * sizeof(float);
    hipMemsetAsync(d_ws, 0, zeroBytes, stream);

    dim3 g1(SLEN / 512, BHNUM);
    cosformer_kv_kernel<<<g1, 256, 0, stream>>>(K, V, mask, kW1, kb1, kW2, kb2, kv, ks);
    dim3 g2(SLEN / 64, BHNUM);
    cosformer_apply_kernel<<<g2, 256, 0, stream>>>(Q, qW1, qb1, qW2, qb2, kv, ks,
                                                   (float*)d_out);
}
